// EdgeCondGCN_7499012899238
// MI455X (gfx1250) — matrix operand order repaired and checked by reading, not compiled, not run on hardware
//
#include <hip/hip_runtime.h>
#include <hip/hip_bf16.h>

typedef __attribute__((ext_vector_type(16))) __bf16 v16bf;
typedef __attribute__((ext_vector_type(8)))  __bf16 v8bf;
typedef __attribute__((ext_vector_type(8)))  float  v8f;

namespace {
constexpr int kBT   = 128;            // B*T
constexpr int kN    = 256;            // nodes
constexpr int kFin  = 32;
constexpr int kFout = 32;
constexpr int kE    = 16;             // edge hidden dim
constexpr int kK1   = kN;             // 256  (contraction j)
constexpr int kKef  = kE * kFin;      // 512  (contraction e*32+f)

// workspace byte offsets
constexpr size_t OFF_HA   = 0;                          // hA  [4096][256] bf16 row-major (2 MB)
constexpr size_t OFF_XB   = (size_t)2u << 20;           // xB  [n=4096][k=256] bf16 col-major (2 MB)
constexpr size_t OFF_WCOL = (size_t)4u << 20;           // Wcol [o=32][ef=512] bf16
constexpr size_t OFF_NW   = ((size_t)4u << 20) + (1u<<16); // NWcol [o=32][f=32] bf16
constexpr size_t OFF_XSUM = ((size_t)4u << 20) + (2u<<16); // xsum [128][32] f32
constexpr size_t OFF_T2   = ((size_t)4u << 20) + (3u<<16); // T2   [128][32] f32
constexpr size_t OFF_S    = (size_t)5u << 20;           // S [i=256][bt=128][ef=512] bf16 (32 MB)
} // namespace

__device__ __forceinline__ float gelu_exact(float v) {
  return 0.5f * v * (1.0f + erff(v * 0.70710678118654752f));
}

__device__ __forceinline__ v8f wmma_bf16(v16bf a, v16bf b, v8f c) {
  return __builtin_amdgcn_wmma_f32_16x16x32_bf16(false, a, false, b, (short)0, c,
                                                 false, false);
}

// A fragment (16x32 bf16, row-major source). Caller passes a per-lane 32-bit
// element offset (row*ld + k0 + half*16); the two 16-byte chunks sit at
// +0 and +8 elements. Uniform base + 32-bit voffset -> saddr addressing.
__device__ __forceinline__ v16bf load_a_frag(const __bf16* __restrict__ base,
                                             unsigned off) {
  v8bf lo = *(const v8bf*)(base + off);
  v8bf hi = *(const v8bf*)(base + off + 8);
  v16bf a;
#pragma unroll
  for (int t = 0; t < 8; ++t) { a[t] = lo[t]; a[8 + t] = hi[t]; }
  return a;
}

// B fragment (32x16 bf16) from column-major staging [col][k]: one contiguous
// 32-byte load per lane (lanes 0-15: K=k0..k0+15, lanes 16-31: K=k0+16..k0+31).
__device__ __forceinline__ v16bf load_b_frag(const __bf16* __restrict__ base,
                                             unsigned off) {
  return *(const v16bf*)(base + off);
}

// ---------------- prep kernels ----------------

// hA[(i*16+e)][j] = bf16( gelu(adj[i,j]*w1[e] + b1[e]) )
__global__ __launch_bounds__(256) void k_build_hA(const float* __restrict__ adj,
                                                  const float* __restrict__ w1,
                                                  const float* __restrict__ b1,
                                                  __bf16* __restrict__ hA) {
  unsigned gid = blockIdx.x * 256 + threadIdx.x;  // 4096*256
  unsigned m = gid >> 8, j = gid & 255;
  unsigned i = m >> 4, e = m & 15;
  float v = adj[i * kN + j] * w1[e] + b1[e];
  hA[m * kK1 + j] = (__bf16)gelu_exact(v);
}

// xB[(bt*32+f)][j] = bf16( x[bt,j,f] )   (column-major over contraction j)
__global__ __launch_bounds__(256) void k_build_xB(const float* __restrict__ x,
                                                  __bf16* __restrict__ xB) {
  unsigned gid = blockIdx.x * 256 + threadIdx.x;  // 4096*256
  unsigned n = gid >> 8, k = gid & 255;
  unsigned bt = n >> 5, f = n & 31;
  xB[n * kK1 + k] = (__bf16)x[(bt * kN + k) * kFin + f];
}

// Wcol[o][e*32+f] = bf16( w2[e][f*32+o] )
__global__ __launch_bounds__(256) void k_build_wcol(const float* __restrict__ w2,
                                                    __bf16* __restrict__ Wcol) {
  unsigned gid = blockIdx.x * 256 + threadIdx.x;  // 32*512
  unsigned o = gid >> 9, ef = gid & 511;
  unsigned e = ef >> 5, f = ef & 31;
  Wcol[gid] = (__bf16)w2[e * (kFin * kFout) + f * kFout + o];
}

// NWcol[o][f] = bf16( node_w[f][o] )
__global__ __launch_bounds__(256) void k_build_nw(const float* __restrict__ node_w,
                                                  __bf16* __restrict__ NWcol) {
  unsigned gid = blockIdx.x * 256 + threadIdx.x;  // 1024
  unsigned o = gid >> 5, f = gid & 31;
  NWcol[gid] = (__bf16)node_w[f * kFout + o];
}

// xsum[bt][f] = sum_j x[bt,j,f]
__global__ __launch_bounds__(256) void k_xsum(const float* __restrict__ x,
                                              float* __restrict__ xsum) {
  unsigned gid = blockIdx.x * 256 + threadIdx.x;  // 128*32
  unsigned bt = gid >> 5, f = gid & 31;
  float s = 0.f;
  for (int j = 0; j < kN; ++j) s += x[(bt * kN + j) * kFin + f];
  xsum[gid] = s;
}

// T2[bt][o] = node_b[o] + sum_f xsum[bt,f]*b2[f*32+o]
__global__ __launch_bounds__(256) void k_t2(const float* __restrict__ xsum,
                                            const float* __restrict__ b2,
                                            const float* __restrict__ node_b,
                                            float* __restrict__ T2) {
  unsigned gid = blockIdx.x * 256 + threadIdx.x;  // 128*32
  unsigned bt = gid >> 5, o = gid & 31;
  float s = node_b[o];
#pragma unroll
  for (int f = 0; f < kFin; ++f) s += xsum[bt * kFin + f] * b2[f * kFout + o];
  T2[gid] = s;
}

// ---------------- GEMM 1: S = hA (4096x256) @ xB (256x4096), bf16 WMMA ----------------
// Output stored permuted: S[i][bt][e*32+f]  (i = m>>4, e = m&15, bt = n>>5, f = n&31)
__global__ __launch_bounds__(256) void k_gemm1(const __bf16* __restrict__ hA,
                                               const __bf16* __restrict__ xB,
                                               __bf16* __restrict__ S) {
  unsigned tid = threadIdx.x;
  unsigned lane = tid & 31, w = tid >> 5;
  unsigned r = lane & 15, hf = lane >> 4;
  unsigned M0 = blockIdx.y * 128 + (w >> 1) * 32;  // this wave: 2 m-tiles
  unsigned N0 = blockIdx.x * 128 + (w & 1) * 64;   // this wave: 4 n-tiles

  // 32-bit per-lane element offsets, computed once; k advances via immediates.
  unsigned offA0 = (M0 + r) * kK1 + hf * 16;
  unsigned offA1 = offA0 + 16u * kK1;
  unsigned offB[4];
#pragma unroll
  for (int nt = 0; nt < 4; ++nt) offB[nt] = (N0 + nt * 16 + r) * kK1 + hf * 16;

  v8f acc[2][4] = {};
#pragma unroll
  for (int ks = 0; ks < kK1 / 32; ++ks) {
    unsigned k0 = ks * 32;
    v16bf a0 = load_a_frag(hA, offA0 + k0);
    v16bf a1 = load_a_frag(hA, offA1 + k0);
    v16bf b[4];
#pragma unroll
    for (int nt = 0; nt < 4; ++nt) b[nt] = load_b_frag(xB, offB[nt] + k0);
#pragma unroll
    for (int nt = 0; nt < 4; ++nt) {
      acc[0][nt] = wmma_bf16(a0, b[nt], acc[0][nt]);
      acc[1][nt] = wmma_bf16(a1, b[nt], acc[1][nt]);
    }
  }

  // store permuted: contiguous-in-f across lanes 0-15 / 16-31 within a tile row
#pragma unroll
  for (int mt = 0; mt < 2; ++mt) {
#pragma unroll
    for (int nt = 0; nt < 4; ++nt) {
      unsigned tm = M0 + mt * 16, tn = N0 + nt * 16;
#pragma unroll
      for (int vg = 0; vg < 8; ++vg) {
        unsigned m = tm + hf * 8 + vg;  // C/D layout: lanes 16-31 -> M += 8
        unsigned n = tn + r;
        unsigned i = m >> 4, e = m & 15, bt = n >> 5, f = n & 31;
        S[(i * kBT + bt) * kKef + e * kFin + f] = (__bf16)acc[mt][nt][vg];
      }
    }
  }
}

// ---------------- GEMM 2 (per node i): agg_i = S_i(128x512) @ W + node term + epilogue
__global__ __launch_bounds__(256) void k_gemm2(const float* __restrict__ x,
                                               const __bf16* __restrict__ S,
                                               const __bf16* __restrict__ Wcol,
                                               const __bf16* __restrict__ NWcol,
                                               const float* __restrict__ T2,
                                               float* __restrict__ out) {
  unsigned i = blockIdx.x;
  unsigned tid = threadIdx.x, lane = tid & 31, w = tid >> 5;
  unsigned r = lane & 15, hf = lane >> 4;
  unsigned btrow = w * 16 + r;  // A-fragment row (bt)

  v8f c[2] = {};

  // node-linear term: x[bt, i, :] (fp32 -> bf16) @ node_w, K = 32 = one WMMA
  {
    unsigned offx = (btrow * kN + i) * kFin + hf * 16;
    v8f lo = *(const v8f*)(x + offx);
    v8f hi = *(const v8f*)(x + offx + 8);
    v16bf a;
#pragma unroll
    for (int t = 0; t < 8; ++t) { a[t] = (__bf16)lo[t]; a[8 + t] = (__bf16)hi[t]; }
#pragma unroll
    for (int nt = 0; nt < 2; ++nt) {
      v16bf b = load_b_frag(NWcol, (nt * 16 + r) * kFin + hf * 16);
      c[nt] = wmma_bf16(a, b, c[nt]);
    }
  }

  // aggregation term: K = 512 over (e,f)
  unsigned offA = i * (kBT * kKef) + btrow * kKef + hf * 16;
  unsigned offW0 = r * kKef + hf * 16;
#pragma unroll
  for (int ks = 0; ks < kKef / 32; ++ks) {
    unsigned k0 = ks * 32;
    v16bf a = load_a_frag(S, offA + k0);
#pragma unroll
    for (int nt = 0; nt < 2; ++nt) {
      v16bf b = load_b_frag(Wcol, offW0 + nt * 16 * kKef + k0);
      c[nt] = wmma_bf16(a, b, c[nt]);
    }
  }

  // epilogue: + T2 (includes node_b), GELU, fp32 store
#pragma unroll
  for (int nt = 0; nt < 2; ++nt) {
#pragma unroll
    for (int vg = 0; vg < 8; ++vg) {
      unsigned bt = w * 16 + hf * 8 + vg;
      unsigned o = nt * 16 + r;
      float v = c[nt][vg] + T2[bt * kFout + o];
      out[(bt * kN + i) * kFout + o] = gelu_exact(v);
    }
  }
}

extern "C" void kernel_launch(void* const* d_in, const int* in_sizes, int n_in,
                              void* d_out, int out_size, void* d_ws, size_t ws_size,
                              hipStream_t stream) {
  const float* x      = (const float*)d_in[0];
  const float* adj    = (const float*)d_in[1];
  const float* w1     = (const float*)d_in[2];
  const float* b1     = (const float*)d_in[3];
  const float* w2     = (const float*)d_in[4];
  const float* b2     = (const float*)d_in[5];
  const float* node_w = (const float*)d_in[6];
  const float* node_b = (const float*)d_in[7];
  float* out = (float*)d_out;

  char* ws = (char*)d_ws;
  __bf16* hA    = (__bf16*)(ws + OFF_HA);
  __bf16* xB    = (__bf16*)(ws + OFF_XB);
  __bf16* Wcol  = (__bf16*)(ws + OFF_WCOL);
  __bf16* NWcol = (__bf16*)(ws + OFF_NW);
  float*  xsum  = (float*)(ws + OFF_XSUM);
  float*  T2    = (float*)(ws + OFF_T2);
  __bf16* S     = (__bf16*)(ws + OFF_S);

  k_build_hA<<<4096, 256, 0, stream>>>(adj, w1, b1, hA);
  k_build_xB<<<4096, 256, 0, stream>>>(x, xB);
  k_build_wcol<<<64, 256, 0, stream>>>(w2, Wcol);
  k_build_nw<<<4, 256, 0, stream>>>(node_w, NWcol);
  k_xsum<<<16, 256, 0, stream>>>(x, xsum);
  k_t2<<<16, 256, 0, stream>>>(xsum, b2, node_b, T2);

  k_gemm1<<<dim3(32, 32), 256, 0, stream>>>(hA, xB, S);
  k_gemm2<<<256, 256, 0, stream>>>(x, S, Wcol, NWcol, T2, out);
}
